// AttentionLayer_39633958207554
// MI455X (gfx1250) — matrix operand order repaired and checked by reading, not compiled, not run on hardware
//
#include <hip/hip_runtime.h>

// ---------------------------------------------------------------------------
// Problem constants (from reference): B=16, T=1024, S=1024, QH=1024, MH=1024
// ---------------------------------------------------------------------------
#define BB  16
#define TT  1024
#define SS  1024
#define QHH 1024
#define MHH 1024

typedef __attribute__((ext_vector_type(16))) __bf16 v16bf;
typedef __attribute__((ext_vector_type(8)))  float  v8f;

union FragAB { v16bf v; uint4 q[2]; };
union FragC  { v8f   v; float f[8]; };

__device__ __forceinline__ unsigned short f2bf(float f) {
    unsigned u = __float_as_uint(f);
    u += 0x7FFFu + ((u >> 16) & 1u);       // round-to-nearest-even
    return (unsigned short)(u >> 16);
}

// ---------------------------------------------------------------------------
// Async global->LDS copy (16B per lane) via gfx1250
// GLOBAL_LOAD_ASYNC_TO_LDS_B128 (typed-pointer builtin:
//   (v4i addrspace(1)*, v4i addrspace(3)*, imm offset, imm cpol)).
// Address-space pointers are materialized through integer casts (a generic
// LDS pointer's low 32 bits == LDS byte offset; a generic global pointer's
// 64-bit value == AS1 address). Falls back to load + ds_store if unavailable.
// ---------------------------------------------------------------------------
typedef int v4i __attribute__((vector_size(16)));
typedef __attribute__((address_space(1))) v4i g_v4i;
typedef __attribute__((address_space(3))) v4i l_v4i;

__device__ __forceinline__ void async_copy16(const unsigned short* g, unsigned short* l) {
#if __has_builtin(__builtin_amdgcn_global_load_async_to_lds_b128)
    __builtin_amdgcn_global_load_async_to_lds_b128(
        (g_v4i*)(unsigned long long)(uintptr_t)g,
        (l_v4i*)(unsigned)(uintptr_t)l,
        0, 0);
#else
    *(uint4*)l = *(const uint4*)g;
#endif
}

__device__ __forceinline__ void async_copy_wait() {
#if __has_builtin(__builtin_amdgcn_global_load_async_to_lds_b128)
#  if __has_builtin(__builtin_amdgcn_s_wait_asynccnt)
    __builtin_amdgcn_s_wait_asynccnt(0);
#  else
    asm volatile("s_wait_asynccnt 0" ::: "memory");
#  endif
#endif
}

// ---------------------------------------------------------------------------
// Elementwise f32 -> bf16 convert (float4 / b128 reads, ushort4 writes)
// ---------------------------------------------------------------------------
__global__ __launch_bounds__(256)
void k_convert_bf16(const float* __restrict__ in, unsigned short* __restrict__ out,
                    long long n4) {
    long long i      = (long long)blockIdx.x * blockDim.x + threadIdx.x;
    long long stride = (long long)gridDim.x * blockDim.x;
    for (; i < n4; i += stride) {
        float4 v = ((const float4*)in)[i];
        ushort4 o;
        o.x = f2bf(v.x); o.y = f2bf(v.y); o.z = f2bf(v.z); o.w = f2bf(v.w);
        ((ushort4*)out)[i] = o;
    }
}

// ---------------------------------------------------------------------------
// memory_bank (B,S,MH) f32  ->  row-major bf16 (B,S,MH)  AND  transposed bf16
// (B,MH,S), via a 32x33 LDS tile (bank-conflict-free transpose).
// ---------------------------------------------------------------------------
__global__ __launch_bounds__(256)
void k_mem_convert_transpose(const float* __restrict__ in,
                             unsigned short* __restrict__ outRM,
                             unsigned short* __restrict__ outTR) {
    __shared__ unsigned short tile[32][33];
    const int b  = blockIdx.z;
    const int s0 = blockIdx.x * 32;
    const int m0 = blockIdx.y * 32;
    const float*    src = in    + (long long)b * SS * MHH;
    unsigned short* drm = outRM + (long long)b * SS * MHH;
    unsigned short* dtr = outTR + (long long)b * MHH * SS;
    const int tx = threadIdx.x & 31;
    const int ty = threadIdx.x >> 5;                 // 0..7
    #pragma unroll
    for (int r = ty; r < 32; r += 8) {
        unsigned short v = f2bf(src[(long long)(s0 + r) * MHH + m0 + tx]);
        tile[r][tx] = v;
        drm[(long long)(s0 + r) * MHH + m0 + tx] = v;
    }
    __syncthreads();
    #pragma unroll
    for (int r = ty; r < 32; r += 8) {
        dtr[(long long)(m0 + r) * SS + s0 + tx] = tile[tx][r];
    }
}

// ---------------------------------------------------------------------------
// Batched C = A(bf16, MxK row-major) @ Bt(bf16, NxK row-major)^T
//
// Block: 256 threads = 8 waves; block tile 128(M) x 64(N).
// Wave w: rows [blockM*128 + w*16, +16), all 64 N columns -> 4 accumulators,
// 4x v_wmma_f32_16x16x32_bf16 per 32-wide k-step, A fragment reused 4x.
// B tile (64 x 32 bf16) staged in double-buffered LDS via async global->LDS
// b128 copies (1 per thread per k-step, issued branch-free with a clamped
// offset and overlapped with the WMMAs); rows padded to 80B so the 16-lane
// ds_load_b128 fragment reads are bank-conflict free (bank stride 20 mod 64).
//
// Fragment addressing matches CDNA5 ISA 7.12.2 VGPR layouts:
//   A  lane: M = row0 + lane%16; 8 bf16 at k+(lane/16)*8 and k+16+(lane/16)*8
//   Bt lane: N = col%16;        8 bf16 at k+(lane/16)*8 and k+16+(lane/16)*8
//   C  vgpr r: (M = row0 + 8*(lane/16) + r, N = col0 + lane%16)
// ---------------------------------------------------------------------------
__global__ __launch_bounds__(256)
void k_gemm_bf16_ABt(const unsigned short* __restrict__ A,
                     const unsigned short* __restrict__ Bt,
                     float* __restrict__ Cf,
                     unsigned short* __restrict__ C16,
                     int N, int K,
                     long long strideA, long long strideB, long long strideC) {
    __shared__ unsigned short ldsB[2][64][40];   // 2 buffers, 64 rows x 32 K (+pad)

    const int b = blockIdx.z;
    A  += (long long)b * strideA;
    Bt += (long long)b * strideB;

    const int tid  = threadIdx.x;
    const int wave = tid >> 5;             // 0..7  -> M sub-tile
    const int lane = tid & 31;
    const int row0 = blockIdx.x * 128 + wave * 16;
    const int col0 = blockIdx.y * 64;

    const int half = lane >> 4;            // 0/1
    const int l16  = lane & 15;

    const unsigned short* aPtr = A + (long long)(row0 + l16) * K + half * 8;

    // cooperative B-tile fill mapping: thread -> (n, k-chunk)
    const int bn = tid >> 2;               // 0..63
    const int bk = (tid & 3) * 8;          // 0,8,16,24
    const unsigned short* bSrc = Bt + (long long)(col0 + bn) * K + bk;
    unsigned short* bDst[2] = { &ldsB[0][bn][bk], &ldsB[1][bn][bk] };

    FragC c[4];
    #pragma unroll
    for (int j = 0; j < 4; j++)
        #pragma unroll
        for (int i = 0; i < 8; i++) c[j].f[i] = 0.0f;

    // preload k-tile 0 into buffer 0
    async_copy16(bSrc, bDst[0]);
    async_copy_wait();
    __syncthreads();

    const int nk = K / 32;
    for (int kt = 0; kt < nk; ++kt) {
        const int k   = kt * 32;
        const int cur = kt & 1;

        // Branch-free prefetch of the next B tile into the other buffer
        // (overlaps with the WMMAs). On the last iteration the clamped offset
        // harmlessly re-fetches the current tile into the never-read buffer.
        const int kpre = (k + 32 < K) ? (k + 32) : k;
        async_copy16(bSrc + kpre, bDst[cur ^ 1]);

        FragAB a;
        a.q[0] = *(const uint4*)(aPtr + k);        // K = k..k+7    (+half*8)
        a.q[1] = *(const uint4*)(aPtr + k + 16);   // K = k+16..23  (+half*8)

        const unsigned short (*bufB)[40] = ldsB[cur];
        #pragma unroll
        for (int j = 0; j < 4; j++) {
            FragAB bf;
            bf.q[0] = *(const uint4*)&bufB[j * 16 + l16][half * 8];
            bf.q[1] = *(const uint4*)&bufB[j * 16 + l16][half * 8 + 16];
            c[j].v = __builtin_amdgcn_wmma_f32_16x16x32_bf16(
                         false, a.v, false, bf.v, (short)0, c[j].v, false, false);
        }

        async_copy_wait();     // our prefetch landed in LDS
        __syncthreads();       // everyone done reading `cur`, writes visible
    }

    const int crow = row0 + half * 8;
    #pragma unroll
    for (int j = 0; j < 4; j++) {
        const int ccol = col0 + j * 16 + l16;
        if (Cf) {
            float* out = Cf + (long long)b * strideC + (long long)crow * N + ccol;
            #pragma unroll
            for (int r = 0; r < 8; r++) out[(long long)r * N] = c[j].f[r];
        } else {
            unsigned short* out = C16 + (long long)b * strideC + (long long)crow * N + ccol;
            #pragma unroll
            for (int r = 0; r < 8; r++) out[(long long)r * N] = f2bf(c[j].f[r]);
        }
    }
}

// ---------------------------------------------------------------------------
// Masked softmax over S=1024, one wave32 per row. In-place normalize the f32
// scores (d_out region) and emit a bf16 copy of P for the context GEMM.
// Positions s >= true_len[b] get probability 0 (reference: -inf before softmax).
// ---------------------------------------------------------------------------
__global__ __launch_bounds__(128)
void k_softmax_rows(float* __restrict__ scores,
                    unsigned short* __restrict__ P16,
                    const long long* __restrict__ true_len) {
    const int row  = blockIdx.x * 4 + (threadIdx.x >> 5);   // 0 .. B*T-1
    const int lane = threadIdx.x & 31;
    const int b    = row / TT;
    const int valid = (int)true_len[b];                     // >= 1 guaranteed

    float* rp = scores + (long long)row * SS;
    float vals[SS / 32];

    float mx = -__builtin_inff();
    #pragma unroll
    for (int i = 0; i < SS / 32; i++) {
        const int s = lane + i * 32;
        float v = (s < valid) ? rp[s] : -__builtin_inff();
        vals[i] = v;
        mx = fmaxf(mx, v);
    }
    #pragma unroll
    for (int off = 16; off >= 1; off >>= 1)
        mx = fmaxf(mx, __shfl_xor(mx, off, 32));

    float sum = 0.0f;
    #pragma unroll
    for (int i = 0; i < SS / 32; i++) {
        const int s = lane + i * 32;
        float e = (s < valid) ? __expf(vals[i] - mx) : 0.0f;
        vals[i] = e;
        sum += e;
    }
    #pragma unroll
    for (int off = 16; off >= 1; off >>= 1)
        sum += __shfl_xor(sum, off, 32);

    const float inv = 1.0f / sum;
    unsigned short* pp = P16 + (long long)row * SS;
    #pragma unroll
    for (int i = 0; i < SS / 32; i++) {
        const int s = lane + i * 32;
        const float p = vals[i] * inv;
        rp[s] = p;
        pp[s] = f2bf(p);
    }
}

// ---------------------------------------------------------------------------
// Host-side launcher
//   d_in[0] = query       (B,T,QH)  f32
//   d_in[1] = memory_bank (B,S,MH)  f32
//   d_in[2] = true_len    (B,)      int64
//   d_in[3] = W_general   (MH,QH)   f32
//   d_out   = context (B*T*MH f32) ++ scores_normalized (B*T*S f32)
// ---------------------------------------------------------------------------
extern "C" void kernel_launch(void* const* d_in, const int* in_sizes, int n_in,
                              void* d_out, int out_size, void* d_ws, size_t ws_size,
                              hipStream_t stream) {
    const float*     query   = (const float*)d_in[0];
    const float*     memory  = (const float*)d_in[1];
    const long long* truelen = (const long long*)d_in[2];
    const float*     W       = (const float*)d_in[3];

    float* context = (float*)d_out;                              // B*T*MH
    float* scores  = (float*)d_out + (size_t)BB * TT * MHH;      // B*T*S

    // Workspace carve-up (all bf16 as raw u16 bits); ~170 MB total.
    char* ws = (char*)d_ws;
    unsigned short* q16  = (unsigned short*)ws; ws += (size_t)BB * TT * QHH * 2;
    unsigned short* w16  = (unsigned short*)ws; ws += (size_t)MHH * QHH * 2;
    unsigned short* m16  = (unsigned short*)ws; ws += (size_t)BB * SS * MHH * 2;
    unsigned short* mT16 = (unsigned short*)ws; ws += (size_t)BB * MHH * SS * 2;
    unsigned short* qp16 = (unsigned short*)ws; ws += (size_t)BB * TT * MHH * 2;
    unsigned short* p16  = (unsigned short*)ws;

    // 1) f32 -> bf16 converts
    k_convert_bf16<<<4096, 256, 0, stream>>>(query, q16, (long long)BB * TT * QHH / 4);
    k_convert_bf16<<<1024, 256, 0, stream>>>(W, w16, (long long)MHH * QHH / 4);
    k_mem_convert_transpose<<<dim3(SS / 32, MHH / 32, BB), 256, 0, stream>>>(
        memory, m16, mT16);

    // 2) q_proj = query @ W^T   (single GEMM: M = B*T = 16384, N = MH, K = QH)
    k_gemm_bf16_ABt<<<dim3((BB * TT) / 128, MHH / 64, 1), 256, 0, stream>>>(
        q16, w16, nullptr, qp16, MHH, QHH, 0, 0, 0);

    // 3) scores = q_proj @ memory^T   (batched: M = T, N = S, K = MH)
    k_gemm_bf16_ABt<<<dim3(TT / 128, SS / 64, BB), 256, 0, stream>>>(
        qp16, m16, scores, nullptr, SS, MHH,
        (long long)TT * MHH, (long long)SS * MHH, (long long)TT * SS);

    // 4) masked softmax (in-place on d_out scores) + bf16 P for next GEMM
    k_softmax_rows<<<(BB * TT) / 4, 128, 0, stream>>>(scores, p16, truelen);

    // 5) context = P @ memory   (via pre-transposed memory: M = T, N = MH, K = S)
    k_gemm_bf16_ABt<<<dim3(TT / 128, MHH / 64, BB), 256, 0, stream>>>(
        p16, mT16, context, nullptr, MHH, SS,
        (long long)TT * SS, (long long)MHH * SS, (long long)TT * MHH);
}
